// LinearSub4Bit_5858335392490
// MI455X (gfx1250) — matrix operand order repaired and checked by reading, not compiled, not run on hardware
//
#include <hip/hip_runtime.h>

// ---------------------------------------------------------------------------
// LinearSub4Bit on MI455X (gfx1250):
//   Phase 1: dequant 3-bit grouped weights -> bf16 W (OUT_F x IN_F) in ws
//            convert x (f32) -> bf16 A (M x IN_F) in ws
//   Phase 2: bf16 WMMA GEMM  C[M,N] = A[M,K] * W[N,K]^T + bias
// M = 2*2048 = 4096, K = 4096, N = 11008.
// Block tile 128(M) x 256(N), 8 waves as 2x4, wave tile 64x64 (4x4 WMMA
// subtiles) -> 16 v_wmma per k-step of 32, 32 FLOP per loaded byte.
// ---------------------------------------------------------------------------

#define IN_F   4096
#define OUT_F  11008
#define M_TOT  4096

typedef __attribute__((ext_vector_type(16))) __bf16 v16bf;
typedef __attribute__((ext_vector_type(8)))  __bf16 v8bf;
typedef __attribute__((ext_vector_type(4)))  __bf16 v4bf;
typedef __attribute__((ext_vector_type(8)))  float  v8f;

// ---------------------- Phase 1a: weight dequant ---------------------------
__global__ __launch_bounds__(256) void dequant_w_kernel(
    const int* __restrict__ q, const float* __restrict__ scales,
    const unsigned char* __restrict__ mask, __bf16* __restrict__ Wbf)
{
    size_t idx = ((size_t)blockIdx.x * blockDim.x + threadIdx.x) * 4;
    float sc = scales[idx >> 6];
    int4 qv = *(const int4*)(q + idx);
    unsigned int mv = *(const unsigned int*)(mask + idx);  // 4 bool bytes
    const float k27 = 2.0f / 7.0f;
    float f0 = ((float)qv.x * k27 - 1.0f) * sc;
    float f1 = ((float)qv.y * k27 - 1.0f) * sc;
    float f2 = ((float)qv.z * k27 - 1.0f) * sc;
    float f3 = ((float)qv.w * k27 - 1.0f) * sc;
    v4bf o;
    o[0] = (mv & 0x000000FFu) ? (__bf16)0.0f : (__bf16)f0;
    o[1] = (mv & 0x0000FF00u) ? (__bf16)0.0f : (__bf16)f1;
    o[2] = (mv & 0x00FF0000u) ? (__bf16)0.0f : (__bf16)f2;
    o[3] = (mv & 0xFF000000u) ? (__bf16)0.0f : (__bf16)f3;
    *(v4bf*)(Wbf + idx) = o;   // 8-byte store
}

// ---------------------- Phase 1b: activation convert -----------------------
__global__ __launch_bounds__(256) void convert_x_kernel(
    const float* __restrict__ x, __bf16* __restrict__ Xbf)
{
    size_t idx = ((size_t)blockIdx.x * blockDim.x + threadIdx.x) * 4;
    float4 v = *(const float4*)(x + idx);
    v4bf o;
    o[0] = (__bf16)v.x; o[1] = (__bf16)v.y;
    o[2] = (__bf16)v.z; o[3] = (__bf16)v.w;
    *(v4bf*)(Xbf + idx) = o;
}

// ---------------------- Phase 2: bf16 WMMA GEMM ----------------------------
// Fragment layouts used here (16-bit, wave32; A and B both k = 16h + s, not the ISA table order):
//   A 16x32: lane holds row m=lane%16; h=lane/16 selects K-halves:
//            V0..3 <- K = 16h..16h+7,  V4..7 <- K = 16h+8..16h+15
//            -> two contiguous 16-byte loads from row-major A.
//   B 32x16: lane holds col n=lane%16; lanes 0-15 K=0..15, lanes 16-31
//            K=16..31 -> one contiguous 32-byte load from row-major W
//            (B = W^T, K contiguous inside each W row).
//   C/D 16x16 f32: element r of v8f = row (r + 8h), col = lane%16.
__global__ __launch_bounds__(256) void gemm_bf16_wmma_kernel(
    const __bf16* __restrict__ A,    // M_TOT x IN_F
    const __bf16* __restrict__ W,    // OUT_F x IN_F
    const float* __restrict__ bias,  // OUT_F
    float* __restrict__ C)           // M_TOT x OUT_F
{
    const int lane = threadIdx.x & 31;
    const int wave = threadIdx.x >> 5;
    const int wm   = wave & 1;    // wave row (M), 0..1
    const int wn   = wave >> 1;   // wave col (N), 0..3
    const int h    = lane >> 4;   // lane half
    const int r16  = lane & 15;

    const int mBase = blockIdx.x * 128 + wm * 64;
    const int nBase = blockIdx.y * 256 + wn * 64;

    v8f acc[4][4] = {};

    // Per-lane stream pointers (element units).
    const __bf16* aRow[4];
#pragma unroll
    for (int mi = 0; mi < 4; ++mi)
        aRow[mi] = A + (size_t)(mBase + mi * 16 + r16) * IN_F + h * 16;
    const __bf16* bRow[4];
#pragma unroll
    for (int ni = 0; ni < 4; ++ni)
        bRow[ni] = W + (size_t)(nBase + ni * 16 + r16) * IN_F + h * 16;

    for (int k = 0; k < IN_F; k += 32) {
        v16bf af[4];
#pragma unroll
        for (int mi = 0; mi < 4; ++mi) {
            v8bf lo = *(const v8bf*)(aRow[mi] + k);        // K = 16h..16h+7
            v8bf hi = *(const v8bf*)(aRow[mi] + k + 8);    // K = 16h+8..16h+15
            af[mi] = __builtin_shufflevector(lo, hi,
                         0,1,2,3,4,5,6,7,8,9,10,11,12,13,14,15);
        }
        v16bf bfrag[4];
#pragma unroll
        for (int ni = 0; ni < 4; ++ni)
            bfrag[ni] = *(const v16bf*)(bRow[ni] + k);     // K = 16h..16h+15

        // Speculative prefetch ~32 k-iterations (2KB/row) ahead; OOB
        // speculative prefetches are silently dropped (ISA 7.3).
#pragma unroll
        for (int mi = 0; mi < 4; ++mi)
            __builtin_prefetch(aRow[mi] + k + 1024, 0, 0);
#pragma unroll
        for (int ni = 0; ni < 4; ++ni)
            __builtin_prefetch(bRow[ni] + k + 1024, 0, 0);

#pragma unroll
        for (int mi = 0; mi < 4; ++mi)
#pragma unroll
            for (int ni = 0; ni < 4; ++ni)
                acc[mi][ni] = __builtin_amdgcn_wmma_f32_16x16x32_bf16(
                    /*neg_a=*/false, af[mi], /*neg_b=*/false, bfrag[ni],
                    /*c_mod=*/(short)0, acc[mi][ni],
                    /*reuse_a=*/false, /*reuse_b=*/false);
    }

    // Fused bias add + store (C/D layout: row = r + 8h, col = r16).
#pragma unroll
    for (int ni = 0; ni < 4; ++ni) {
        const int col = nBase + ni * 16 + r16;
        const float bv = bias[col];
#pragma unroll
        for (int mi = 0; mi < 4; ++mi) {
#pragma unroll
            for (int r = 0; r < 8; ++r) {
                const int row = mBase + mi * 16 + h * 8 + r;
                C[(size_t)row * OUT_F + col] = acc[mi][ni][r] + bv;
            }
        }
    }
}

// ---------------------------------------------------------------------------
extern "C" void kernel_launch(void* const* d_in, const int* in_sizes, int n_in,
                              void* d_out, int out_size, void* d_ws, size_t ws_size,
                              hipStream_t stream) {
    const float*         x      = (const float*)d_in[0];          // (2,2048,4096) f32
    const int*           q      = (const int*)d_in[1];            // (NG,64) int32 in [0,8)
    const float*         scales = (const float*)d_in[2];          // (NG,) f32
    const unsigned char* mask   = (const unsigned char*)d_in[3];  // (NG,64) bool
    const float*         bias   = (const float*)d_in[4];          // (11008,) f32
    float*               out    = (float*)d_out;                  // (2,2048,11008) f32

    // Workspace layout: [ Xbf: M_TOT*IN_F bf16 | Wbf: OUT_F*IN_F bf16 ]
    __bf16* Xbf = (__bf16*)d_ws;
    __bf16* Wbf = Xbf + (size_t)M_TOT * IN_F;

    // Phase 1b: x -> bf16 (16,777,216 elems / 4 per thread / 256 per block)
    convert_x_kernel<<<(M_TOT * (size_t)IN_F) / (4 * 256), 256, 0, stream>>>(x, Xbf);

    // Phase 1a: weights -> bf16 (45,088,768 elems / 4 per thread)
    dequant_w_kernel<<<((size_t)OUT_F * IN_F) / (4 * 256), 256, 0, stream>>>(
        q, scales, mask, Wbf);

    // Phase 2: GEMM, grid = (4096/128, 11008/256) = (32, 43)
    dim3 grid(M_TOT / 128, OUT_F / 256);
    gemm_bf16_wmma_kernel<<<grid, 256, 0, stream>>>(Xbf, Wbf, bias, out);
}
